// MABPostLN_40870908789051
// MI455X (gfx1250) — matrix operand order repaired and checked by reading, not compiled, not run on hardware
//
#include <hip/hip_runtime.h>
#include <hip/hip_bf16.h>
#include <math.h>

typedef __attribute__((ext_vector_type(16))) _Float16 v16h;
typedef __attribute__((ext_vector_type(8)))  _Float16 v8h;
typedef __attribute__((ext_vector_type(8)))  float    v8f;

#define Dm   1024
#define Hh   16
#define HD   64
#define Bb   8
#define SQ   1024
#define MTOT (Bb*SQ)

union Frag16 { v16h v; v8h h[2]; };

static __device__ __forceinline__ v8f wmma_f16(const Frag16& a, const Frag16& b, v8f c) {
    // 8 args: (neg_a, A, neg_b, B, c_mod, C, reuse_a, reuse_b)
    return __builtin_amdgcn_wmma_f32_16x16x32_f16(false, a.v, false, b.v, (short)0, c, false, false);
}

// LDS byte offset of a __shared__ object: generic LDS addresses carry the LDS
// offset in addr[31:0] (aperture truncation semantics).
static __device__ __forceinline__ unsigned lds_off(const void* p) {
    return (unsigned)(unsigned long long)(uintptr_t)p;
}

// CDNA5 async copy global -> LDS (16B per lane), tracked by ASYNCcnt.
static __device__ __forceinline__ void async_g2l_b128(unsigned lds_dst, const void* gsrc) {
    asm volatile("global_load_async_to_lds_b128 %0, %1, off"
                 :: "v"(lds_dst), "v"((unsigned long long)(uintptr_t)gsrc)
                 : "memory");
}
static __device__ __forceinline__ void wait_async0() {
    asm volatile("s_wait_asynccnt 0x0" ::: "memory");
}

// ---------------------------------------------------------------------------
// fp32 -> f16 conversion (vectorized by 4)
// ---------------------------------------------------------------------------
__global__ __launch_bounds__(256)
void cvt_f32_f16_kernel(const float* __restrict__ in, _Float16* __restrict__ out, int n4) {
    int i = blockIdx.x * 256 + threadIdx.x;
    if (i < n4) {
        float4 v = reinterpret_cast<const float4*>(in)[i];
        _Float16* o = out + (size_t)i * 4;
        o[0] = (_Float16)v.x; o[1] = (_Float16)v.y;
        o[2] = (_Float16)v.z; o[3] = (_Float16)v.w;
    }
}

// ---------------------------------------------------------------------------
// GEMM: C[M,1024] = A[M,1024](f16) @ W[1024,1024](f16) + bias, optional GELU.
// Block tile 128x128, K-step 32, 8 waves; wave computes 32(M) x 64(N).
// Double-buffered: A tile via GLOBAL_LOAD_ASYNC_TO_LDS_B128 issued for tile
// i+1 before computing tile i, so the async DMA overlaps the WMMA phase.
// ---------------------------------------------------------------------------
__global__ __launch_bounds__(256)
void gemm_bias_kernel(const _Float16* __restrict__ A,
                      const _Float16* __restrict__ W,
                      const float*    __restrict__ bias,
                      float*          __restrict__ outF,   // nullable
                      _Float16*       __restrict__ outH,   // nullable
                      int gelu)
{
    __shared__ _Float16 ldsA[2][128 * 40];   // row-major, stride 40 halves
    __shared__ _Float16 ldsB[2][128 * 40];   // column-major W^T: [n][k]

    const int t    = threadIdx.x;
    const int lane = t & 31;
    const int wv   = t >> 5;
    const int wm   = wv >> 1;     // 0..3
    const int wn   = wv & 1;      // 0..1
    const int ln16 = lane & 15;
    const int hi   = lane >> 4;

    const int m0 = blockIdx.y * 128;
    const int n0 = blockIdx.x * 128;

    v8f acc[2][4];
#pragma unroll
    for (int mt = 0; mt < 2; ++mt)
#pragma unroll
        for (int j = 0; j < 4; ++j) acc[mt][j] = (v8f){};

    // staging assignments
    const int ar = t >> 1;             // A row 0..127
    const int ac = (t & 1) * 16;       // A col-half
    const int bk = t & 31;             // B k-row 0..31
    const int bn = (t >> 5) * 16;      // B n-base
    const unsigned dstA[2] = { lds_off(&ldsA[0][ar * 40 + ac]),
                               lds_off(&ldsA[1][ar * 40 + ac]) };
    const _Float16* gArow = A + (size_t)(m0 + ar) * 1024 + ac;
    const _Float16* gBrow = W + (size_t)bk * 1024 + n0 + bn;

    // ---- prologue: stage tile 0 into buffer 0
    {
        async_g2l_b128(dstA[0],      gArow);
        async_g2l_b128(dstA[0] + 16, gArow + 8);
        v8h w0 = *reinterpret_cast<const v8h*>(gBrow);
        v8h w1 = *reinterpret_cast<const v8h*>(gBrow + 8);
#pragma unroll
        for (int i = 0; i < 8; ++i) {
            ldsB[0][(bn + i)     * 40 + bk] = w0[i];
            ldsB[0][(bn + i + 8) * 40 + bk] = w1[i];
        }
        wait_async0();
        __syncthreads();
    }

#pragma unroll 2
    for (int it = 0; it < 32; ++it) {
        const int p  = it & 1;
        const int np = p ^ 1;
        const bool pf = (it + 1) < 32;
        v8h w0 = {}, w1 = {};
        if (pf) {
            const int kk = (it + 1) * 32;
            // next A tile: async DMA straight into the other LDS buffer
            async_g2l_b128(dstA[np],      gArow + kk);
            async_g2l_b128(dstA[np] + 16, gArow + kk + 8);
            // next B tile: global loads now, LDS scatter after compute
            w0 = *reinterpret_cast<const v8h*>(gBrow + (size_t)kk * 1024);
            w1 = *reinterpret_cast<const v8h*>(gBrow + (size_t)kk * 1024 + 8);
        }

        // ---- compute on buffer p
        Frag16 a[2];
#pragma unroll
        for (int mt = 0; mt < 2; ++mt) {
            const int row = wm * 32 + mt * 16 + ln16;
            a[mt].h[0] = *reinterpret_cast<const v8h*>(&ldsA[p][row * 40 + hi * 8]);
            a[mt].h[1] = *reinterpret_cast<const v8h*>(&ldsA[p][row * 40 + 16 + hi * 8]);
        }
#pragma unroll
        for (int j = 0; j < 4; ++j) {
            const int col = wn * 64 + j * 16 + ln16;
            Frag16 b;
            b.h[0] = *reinterpret_cast<const v8h*>(&ldsB[p][col * 40 + hi * 8]);
            b.h[1] = *reinterpret_cast<const v8h*>(&ldsB[p][col * 40 + 16 + hi * 8]);
#pragma unroll
            for (int mt = 0; mt < 2; ++mt)
                acc[mt][j] = wmma_f16(a[mt], b, acc[mt][j]);
        }

        // ---- finish staging tile i+1
        if (pf) {
#pragma unroll
            for (int i = 0; i < 8; ++i) {
                ldsB[np][(bn + i)     * 40 + bk] = w0[i];
                ldsB[np][(bn + i + 8) * 40 + bk] = w1[i];
            }
            wait_async0();
        }
        __syncthreads();
    }

    // epilogue: bias (+GELU) and store
#pragma unroll
    for (int j = 0; j < 4; ++j) {
        const int col = n0 + wn * 64 + j * 16 + ln16;
        const float bv = bias[col];
#pragma unroll
        for (int mt = 0; mt < 2; ++mt) {
            const int rbase = m0 + wm * 32 + mt * 16 + hi * 8;
#pragma unroll
            for (int r = 0; r < 8; ++r) {
                float x = acc[mt][j][r] + bv;
                if (gelu) x = 0.5f * x * (1.0f + erff(x * 0.70710678118f));
                const size_t idx = (size_t)(rbase + r) * 1024 + col;
                if (outF) outF[idx] = x;
                if (outH) outH[idx] = (_Float16)x;
            }
        }
    }
}

// ---------------------------------------------------------------------------
// Flash attention per (b,h): 4 waves/block, each wave owns 16 q-rows, kv tiles
// of 64. Row stats stay within one 16-lane group -> shuffle-only reductions.
// Double-buffered K (async global->LDS) and V (manual transpose) staging.
// ---------------------------------------------------------------------------
__global__ __launch_bounds__(128)
void attn_kernel(const _Float16* __restrict__ q,    // [B,Sq,D] f16
                 const _Float16* __restrict__ k,
                 const _Float16* __restrict__ v,
                 const int*      __restrict__ mask,  // [B,Skv]
                 float*          __restrict__ ctx)   // [B,Sq,D] f32
{
    __shared__ _Float16 ldsK[2][64 * 72];    // [kv][d], stride 72 halves
    __shared__ _Float16 ldsV[2][64 * 72];    // [d][kv] (V transposed)
    __shared__ _Float16 ldsP[4 * 16 * 72];   // per-wave P tile [m][kv]

    const int t    = threadIdx.x;
    const int lane = t & 31;
    const int wv   = t >> 5;
    const int ln16 = lane & 15;
    const int hi   = lane >> 4;
    const int b    = blockIdx.y >> 4;
    const int h    = blockIdx.y & 15;
    const int q0   = blockIdx.x * 64 + wv * 16;

    // Q A-fragments, loaded once straight from global
    Frag16 aq[2];
    {
        const _Float16* gq = q + ((size_t)(b * SQ + q0 + ln16)) * Dm + h * HD;
#pragma unroll
        for (int kt = 0; kt < 2; ++kt) {
            aq[kt].h[0] = *reinterpret_cast<const v8h*>(gq + kt * 32 + hi * 8);
            aq[kt].h[1] = *reinterpret_cast<const v8h*>(gq + kt * 32 + 16 + hi * 8);
        }
    }

    float mrow[8], lrow[8];
    v8f o[4];
#pragma unroll
    for (int r = 0; r < 8; ++r) { mrow[r] = -3.0e38f; lrow[r] = 0.0f; }
#pragma unroll
    for (int j = 0; j < 4; ++j) o[j] = (v8f){};

    const int skv   = t >> 1;          // staging: kv row 0..63
    const int chalf = (t & 1) * 32;    // staging: d half 0/32
    const unsigned dstK[2] = { lds_off(&ldsK[0][skv * 72 + chalf]),
                               lds_off(&ldsK[1][skv * 72 + chalf]) };
    const _Float16* gkRow = k + ((size_t)(b * SQ + skv)) * Dm + h * HD + chalf;
    const _Float16* gvRow = v + ((size_t)(b * SQ + skv)) * Dm + h * HD + chalf;

    // ---- prologue: stage kv tile 0 into buffer 0
    {
#pragma unroll
        for (int i = 0; i < 4; ++i)
            async_g2l_b128(dstK[0] + i * 16, gkRow + i * 8);
#pragma unroll
        for (int i = 0; i < 4; ++i) {
            v8h x = *reinterpret_cast<const v8h*>(gvRow + i * 8);
#pragma unroll
            for (int jj = 0; jj < 8; ++jj)
                ldsV[0][(chalf + i * 8 + jj) * 72 + skv] = x[jj];
        }
        wait_async0();
        __syncthreads();
    }

#pragma unroll 2
    for (int it = 0; it < 16; ++it) {
        const int p  = it & 1;
        const int np = p ^ 1;
        const bool pf = (it + 1) < 16;
        const int kv0 = it * 64;

        v8h vx[4] = {};
        if (pf) {
            const size_t nxt = (size_t)(it + 1) * 64 * Dm;
#pragma unroll
            for (int i = 0; i < 4; ++i)
                async_g2l_b128(dstK[np] + i * 16, gkRow + nxt + i * 8);
#pragma unroll
            for (int i = 0; i < 4; ++i)
                vx[i] = *reinterpret_cast<const v8h*>(gvRow + nxt + i * 8);
        }

        // ---- S = Q K^T  (per wave: 16 x 64) on buffer p
        v8f s[4];
#pragma unroll
        for (int j = 0; j < 4; ++j) {
            s[j] = (v8f){};
            const int kvc = j * 16 + ln16;
            Frag16 bk_;
#pragma unroll
            for (int kt = 0; kt < 2; ++kt) {
                bk_.h[0] = *reinterpret_cast<const v8h*>(&ldsK[p][kvc * 72 + kt * 32 + hi * 8]);
                bk_.h[1] = *reinterpret_cast<const v8h*>(&ldsK[p][kvc * 72 + kt * 32 + 16 + hi * 8]);
                s[j] = wmma_f16(aq[kt], bk_, s[j]);
            }
            // scale + key padding mask (column = kv0 + j*16 + ln16)
            const int mv = mask[b * SQ + kv0 + j * 16 + ln16];
#pragma unroll
            for (int r = 0; r < 8; ++r)
                s[j][r] = mv ? s[j][r] * 0.125f : -3.0e38f;
        }

        // ---- online softmax (per row, within 16-lane group)
#pragma unroll
        for (int r = 0; r < 8; ++r) {
            float tm = fmaxf(fmaxf(s[0][r], s[1][r]), fmaxf(s[2][r], s[3][r]));
#pragma unroll
            for (int msk = 1; msk < 16; msk <<= 1)
                tm = fmaxf(tm, __shfl_xor(tm, msk, 32));
            const float nm = fmaxf(mrow[r], tm);
            const float sc = __expf(mrow[r] - nm);
            mrow[r] = nm;
            float ps = 0.0f;
#pragma unroll
            for (int j = 0; j < 4; ++j) {
                const float pe = __expf(s[j][r] - nm);
                s[j][r] = pe;
                ps += pe;
            }
#pragma unroll
            for (int msk = 1; msk < 16; msk <<= 1)
                ps += __shfl_xor(ps, msk, 32);
            lrow[r] = lrow[r] * sc + ps;
#pragma unroll
            for (int j = 0; j < 4; ++j) o[j][r] *= sc;
        }

        // ---- P (D-layout) -> per-wave LDS -> A-fragment layout
        _Float16* pw = &ldsP[wv * 16 * 72];
#pragma unroll
        for (int j = 0; j < 4; ++j)
#pragma unroll
            for (int r = 0; r < 8; ++r)
                pw[(hi * 8 + r) * 72 + j * 16 + ln16] = (_Float16)s[j][r];

        Frag16 ap[2];
#pragma unroll
        for (int kt = 0; kt < 2; ++kt) {
            ap[kt].h[0] = *reinterpret_cast<const v8h*>(&pw[ln16 * 72 + kt * 32 + hi * 8]);
            ap[kt].h[1] = *reinterpret_cast<const v8h*>(&pw[ln16 * 72 + kt * 32 + 16 + hi * 8]);
        }

        // ---- O += P V on buffer p
#pragma unroll
        for (int j = 0; j < 4; ++j) {
            const int dc = j * 16 + ln16;
            Frag16 bv_;
#pragma unroll
            for (int kt = 0; kt < 2; ++kt) {
                bv_.h[0] = *reinterpret_cast<const v8h*>(&ldsV[p][dc * 72 + kt * 32 + hi * 8]);
                bv_.h[1] = *reinterpret_cast<const v8h*>(&ldsV[p][dc * 72 + kt * 32 + 16 + hi * 8]);
                o[j] = wmma_f16(ap[kt], bv_, o[j]);
            }
        }

        // ---- finish staging next V tile, publish buffers
        if (pf) {
#pragma unroll
            for (int i = 0; i < 4; ++i)
#pragma unroll
                for (int jj = 0; jj < 8; ++jj)
                    ldsV[np][(chalf + i * 8 + jj) * 72 + skv] = vx[i][jj];
            wait_async0();
        }
        __syncthreads();
    }

    // normalize and scatter ctx
#pragma unroll
    for (int r = 0; r < 8; ++r) {
        const float inv = (lrow[r] > 0.0f) ? (1.0f / lrow[r]) : 0.0f;
        const size_t row = (size_t)(b * SQ + q0 + hi * 8 + r);
#pragma unroll
        for (int j = 0; j < 4; ++j)
            ctx[row * Dm + h * HD + j * 16 + ln16] = o[j][r] * inv;
    }
}

// ---------------------------------------------------------------------------
// out = LayerNorm(a + b) * g + beta      (row-wise over D=1024)
// ---------------------------------------------------------------------------
__global__ __launch_bounds__(256)
void ln_kernel(const float* __restrict__ a, const float* __restrict__ bres,
               const float* __restrict__ g, const float* __restrict__ beta,
               float* __restrict__ outF, _Float16* __restrict__ outH)
{
    __shared__ float redS[8];
    __shared__ float redQ[8];
    const int t = threadIdx.x;
    const int lane = t & 31, wid = t >> 5;
    const size_t row = blockIdx.x;

    const float4 av = reinterpret_cast<const float4*>(a    + row * 1024)[t];
    const float4 bv = reinterpret_cast<const float4*>(bres + row * 1024)[t];
    float xs[4] = { av.x + bv.x, av.y + bv.y, av.z + bv.z, av.w + bv.w };

    float s = 0.0f, s2 = 0.0f;
#pragma unroll
    for (int i = 0; i < 4; ++i) { s += xs[i]; s2 += xs[i] * xs[i]; }
#pragma unroll
    for (int msk = 1; msk < 32; msk <<= 1) {
        s  += __shfl_xor(s,  msk, 32);
        s2 += __shfl_xor(s2, msk, 32);
    }
    if (lane == 0) { redS[wid] = s; redQ[wid] = s2; }
    __syncthreads();
    if (t == 0) {
        float a0 = 0.0f, a1 = 0.0f;
#pragma unroll
        for (int i = 0; i < 8; ++i) { a0 += redS[i]; a1 += redQ[i]; }
        redS[0] = a0; redQ[0] = a1;
    }
    __syncthreads();
    const float mean = redS[0] * (1.0f / 1024.0f);
    const float var  = redQ[0] * (1.0f / 1024.0f) - mean * mean;
    const float rstd = rsqrtf(var + 1e-5f);

#pragma unroll
    for (int i = 0; i < 4; ++i) {
        const int c = t * 4 + i;
        const float y = (xs[i] - mean) * rstd * g[c] + beta[c];
        outF[row * 1024 + c] = y;
        if (outH) outH[row * 1024 + c] = (_Float16)y;
    }
}

// ---------------------------------------------------------------------------
extern "C" void kernel_launch(void* const* d_in, const int* in_sizes, int n_in,
                              void* d_out, int out_size, void* d_ws, size_t ws_size,
                              hipStream_t stream) {
    const float* Q    = (const float*)d_in[0];
    const float* K    = (const float*)d_in[1];
    const int*   mask = (const int*)  d_in[2];
    const float* Wq   = (const float*)d_in[3];
    const float* bq   = (const float*)d_in[4];
    const float* Wk   = (const float*)d_in[5];
    const float* bk   = (const float*)d_in[6];
    const float* Wv   = (const float*)d_in[7];
    const float* bv   = (const float*)d_in[8];
    const float* Wp   = (const float*)d_in[9];
    const float* bp   = (const float*)d_in[10];
    const float* g0   = (const float*)d_in[11];
    const float* be0  = (const float*)d_in[12];
    const float* g1   = (const float*)d_in[13];
    const float* be1  = (const float*)d_in[14];

    char* ws = (char*)d_ws;
    size_t off = 0;
    auto allocH = [&](size_t n) { _Float16* p = (_Float16*)(ws + off); off += n * 2; return p; };
    auto allocF = [&](size_t n) { float*    p = (float*)   (ws + off); off += n * 4; return p; };
    const size_t ACT = (size_t)MTOT * Dm;   // 8M elems
    const size_t WEL = (size_t)Dm * Dm;     // 1M elems

    _Float16* Qh  = allocH(ACT);
    _Float16* Kh  = allocH(ACT);
    _Float16* Wqh = allocH(WEL);
    _Float16* Wkh = allocH(WEL);
    _Float16* Wvh = allocH(WEL);
    _Float16* Wph = allocH(WEL);
    _Float16* qh  = allocH(ACT);
    _Float16* kh  = allocH(ACT);
    _Float16* vh  = allocH(ACT);
    _Float16* xh  = allocH(ACT);
    float*    qf  = allocF(ACT);
    float*    ctx = allocF(ACT);
    float*    x   = allocF(ACT);
    float*    gy  = ctx;    // ctx is dead after LN1; reuse for GELU-GEMM output

    // fp32 -> f16 conversions
    cvt_f32_f16_kernel<<<(int)(ACT / 1024), 256, 0, stream>>>(Q, Qh, (int)(ACT / 4));
    cvt_f32_f16_kernel<<<(int)(ACT / 1024), 256, 0, stream>>>(K, Kh, (int)(ACT / 4));
    cvt_f32_f16_kernel<<<(int)(WEL / 1024), 256, 0, stream>>>(Wq, Wqh, (int)(WEL / 4));
    cvt_f32_f16_kernel<<<(int)(WEL / 1024), 256, 0, stream>>>(Wk, Wkh, (int)(WEL / 4));
    cvt_f32_f16_kernel<<<(int)(WEL / 1024), 256, 0, stream>>>(Wv, Wvh, (int)(WEL / 4));
    cvt_f32_f16_kernel<<<(int)(WEL / 1024), 256, 0, stream>>>(Wp, Wph, (int)(WEL / 4));

    const dim3 gg(Dm / 128, MTOT / 128);   // 8 x 64
    // q = Q@Wq + bq   (keep fp32 for residual, f16 for attention)
    gemm_bias_kernel<<<gg, 256, 0, stream>>>(Qh, Wqh, bq, qf, qh, 0);
    // k = K@Wk + bk ; v = K@Wv + bv
    gemm_bias_kernel<<<gg, 256, 0, stream>>>(Kh, Wkh, bk, nullptr, kh, 0);
    gemm_bias_kernel<<<gg, 256, 0, stream>>>(Kh, Wvh, bv, nullptr, vh, 0);

    // ctx = softmax(q k^T / 8, mask) v
    attn_kernel<<<dim3(SQ / 64, Bb * Hh), 128, 0, stream>>>(qh, kh, vh, mask, ctx);

    // x = LN(q + ctx)
    ln_kernel<<<MTOT, 256, 0, stream>>>(qf, ctx, g0, be0, x, xh);

    // gy = gelu(x@Wp + bp)
    gemm_bias_kernel<<<gg, 256, 0, stream>>>(xh, Wph, bp, gy, nullptr, 1);

    // out = LN(x + gy)
    ln_kernel<<<MTOT, 256, 0, stream>>>(x, gy, g1, be1, (float*)d_out, nullptr);
}
